// CAE_DG_pair_75222057222833
// MI455X (gfx1250) — matrix operand order repaired and checked by reading, compiled, not run on hardware
//
#include <hip/hip_runtime.h>
#include <hip/hip_bf16.h>
#include <math.h>

typedef __attribute__((ext_vector_type(16))) __bf16 v16bf;
typedef __attribute__((ext_vector_type(8)))  float  v8f;

static inline int cdiv_h(int a, int b) { return (a + b - 1) / b; }

__device__ inline float atomicMaxFloat(float* addr, float val) {
  int* ia = (int*)addr;
  int old = __float_as_int(*addr);
  while (__int_as_float(old) < val) {
    int assumed = old;
    old = atomicCAS(ia, assumed, __float_as_int(val));
    if (old == assumed) break;
  }
  return __int_as_float(old);
}

// ---------------------------------------------------------------------------
// Generic direct convolution (NCHW) + leaky ReLU
// ---------------------------------------------------------------------------
__global__ __launch_bounds__(256)
void conv2d_leaky_kernel(const float* __restrict__ in, const float* __restrict__ w,
                         const float* __restrict__ bias, float* __restrict__ out,
                         int B, int Cin, int H, int W, int Cout,
                         int KH, int KW, int stride, int pad, int OH, int OW, float slope) {
  long long t = (long long)blockIdx.x * blockDim.x + threadIdx.x;
  long long total = (long long)B * Cout * OH * OW;
  if (t >= total) return;
  int ox = (int)(t % OW);
  int oy = (int)((t / OW) % OH);
  int oc = (int)((t / ((long long)OW * OH)) % Cout);
  int b  = (int)(t / ((long long)OW * OH * Cout));
  float s = bias[oc];
  for (int ic = 0; ic < Cin; ++ic) {
    for (int ky = 0; ky < KH; ++ky) {
      int iy = oy * stride - pad + ky;
      if (iy < 0 || iy >= H) continue;
      for (int kx = 0; kx < KW; ++kx) {
        int ix = ox * stride - pad + kx;
        if (ix < 0 || ix >= W) continue;
        s += in[(((long long)b * Cin + ic) * H + iy) * W + ix] *
             w[(((long long)oc * Cin + ic) * KH + ky) * KW + kx];
      }
    }
  }
  s = (s >= 0.f) ? s : slope * s;
  out[t] = s;
}

// ---------------------------------------------------------------------------
// xx[b,n] = sum_c f[b,c,n]^2
// ---------------------------------------------------------------------------
__global__ __launch_bounds__(256)
void xx_kernel(const float* __restrict__ fin, float* __restrict__ xx, int B, int C, int Np) {
  int t = blockIdx.x * blockDim.x + threadIdx.x;
  if (t >= B * Np) return;
  int b = t / Np, n = t % Np;
  float s = 0.f;
  for (int c = 0; c < C; ++c) {
    float v = fin[((size_t)b * C + c) * Np + n];
    s += v * v;
  }
  xx[t] = s;
}

// ---------------------------------------------------------------------------
// Fused pairwise-distance + top-16 neighbor search (one thread per point).
// dist(n,m) = 2 * <f_n, f_m> - ||f_m||^2   (row-constant -||f_n||^2 dropped;
// ordering and tie-breaking (lowest index first) match lax.top_k)
// ---------------------------------------------------------------------------
__global__ __launch_bounds__(32)
void topk_kernel(const float* __restrict__ fin, const float* __restrict__ xx,
                 int* __restrict__ idxOut, int C, int Np) {
  __shared__ float ctr[32][257];
  int n = blockIdx.x * 32 + threadIdx.x;
  int b = blockIdx.y;
  for (int c = 0; c < C; ++c)
    ctr[threadIdx.x][c] = fin[((size_t)b * C + c) * Np + n];
  float val[16];
  int   id[16];
#pragma unroll
  for (int i = 0; i < 16; ++i) { val[i] = -3.0e38f; id[i] = 0; }
  for (int m = 0; m < Np; ++m) {
    float dot = 0.f;
    for (int c = 0; c < C; ++c)
      dot += ctr[threadIdx.x][c] * fin[((size_t)b * C + c) * Np + m];
    float dist = 2.f * dot - xx[(size_t)b * Np + m];
    if (dist > val[15]) {
      val[15] = dist; id[15] = m;
#pragma unroll
      for (int t = 15; t > 0; --t) {
        if (val[t] > val[t - 1]) {
          float tv = val[t]; val[t] = val[t - 1]; val[t - 1] = tv;
          int   ti = id[t];  id[t]  = id[t - 1];  id[t - 1]  = ti;
        }
      }
    }
  }
  for (int i = 0; i < 16; ++i)
    idxOut[((size_t)b * Np + n) * 16 + i] = id[i];
}

// ---------------------------------------------------------------------------
// Build edge-feature B matrix for one batch: featB[(2C) x (2048*16)] bf16
// cp < C : f[b,cp,idx] - f[b,cp,n] ; cp >= C : f[b,cp-C,n]
// ---------------------------------------------------------------------------
__global__ __launch_bounds__(256)
void build_feat_kernel(const float* __restrict__ fin, const int* __restrict__ idx,
                       __bf16* __restrict__ featB, int b, int C, int Np) {
  int j  = blockIdx.x * blockDim.x + threadIdx.x;   // 0..Np*16-1
  int cp = blockIdx.y;                              // 0..2C-1
  int NJ = Np * 16;
  if (j >= NJ) return;
  int n = j >> 4, k = j & 15;
  int m = idx[((size_t)b * Np + n) * 16 + k];
  float v;
  if (cp < C)
    v = fin[((size_t)b * C + cp) * Np + m] - fin[((size_t)b * C + cp) * Np + n];
  else
    v = fin[((size_t)b * C + (cp - C)) * Np + n];
  featB[(size_t)cp * NJ + j] = (__bf16)v;
}

// ---------------------------------------------------------------------------
// Build fc = concat(g1..g4) B matrix for one batch: fcB[512 x 2048] bf16
// ---------------------------------------------------------------------------
__global__ __launch_bounds__(256)
void build_fc_kernel(const float* __restrict__ g1, const float* __restrict__ g2,
                     const float* __restrict__ g3, const float* __restrict__ g4,
                     __bf16* __restrict__ fcB, int b) {
  int n = blockIdx.x * blockDim.x + threadIdx.x;   // 0..2047
  int c = blockIdx.y;                              // 0..511
  if (n >= 2048) return;
  float v;
  if (c < 64)       v = g1[((size_t)b * 64  + c)         * 2048 + n];
  else if (c < 128) v = g2[((size_t)b * 64  + (c - 64))  * 2048 + n];
  else if (c < 256) v = g3[((size_t)b * 128 + (c - 128)) * 2048 + n];
  else              v = g4[((size_t)b * 256 + (c - 256)) * 2048 + n];
  fcB[(size_t)c * 2048 + n] = (__bf16)v;
}

// ---------------------------------------------------------------------------
// WMMA GEMM: D = A (MxK fp32, converted to bf16 on the fly) x B (KxN bf16)
// - B tile stored transposed in LDS -> fragment reads are 2x ds_load_b128
// - 80B LDS row stride: 16B aligned + conflict-free lane stride
// - register-staged double buffering: next tile's global loads are issued
//   before the current tile's WMMA sequence (latency overlap)
// - fast path (whole tile in-bounds, aligned): float4/uint4 vector loads,
//   branch-free; slow path only for edge tiles / tiny GEMMs
// mode 0: D[row*N+col] = acc (+bias[row])
// mode 1: per-channel sum / sumsq accumulation (BN statistics)
// mode 2: BN apply + leaky(0.2) + max over k (16-col groups) -> gout[b,ch,n]
// mode 3: BN apply + leaky(0.2) + fused max/mean pooling -> pool[b, 2048]
// ---------------------------------------------------------------------------
__global__ __launch_bounds__(128)
void gemm_bf16_wmma(const float* __restrict__ A, const __bf16* __restrict__ B,
                    int M, int N, int K, int mode,
                    float* __restrict__ D, const float* __restrict__ bias,
                    float* __restrict__ statSum, float* __restrict__ statSsq,
                    const float* __restrict__ scaleV, const float* __restrict__ shiftV,
                    float* __restrict__ gout, int batch, int gldN,
                    float* __restrict__ pool) {
  __shared__ __bf16 As[2][64][40];    // As[buf][row][k]
  __shared__ __bf16 BsT[2][64][40];   // BsT[buf][col][k]
  const int tid  = threadIdx.x;
  const int wave = tid >> 5;
  const int lane = tid & 31;
  const int half = lane >> 4;
  const int ln   = lane & 15;
  const int tileM = blockIdx.y * 64;
  const int tileN = blockIdx.x * 64;

  // staging coordinates
  const int arow = tid >> 1;           // 0..63
  const int acol = (tid & 1) << 4;     // 0 / 16
  const int brow = tid >> 2;           // 0..31 (k-row)
  const int bcol = (tid & 3) << 4;     // 0 / 16 / 32 / 48

  const bool aRowOk   = (tileM + 64 <= M) && ((K & 3) == 0);
  const bool bColOk   = (tileN + 64 <= N) && ((N & 7) == 0);

  v8f acc[4];
#pragma unroll
  for (int t = 0; t < 4; ++t)
#pragma unroll
    for (int j = 0; j < 8; ++j) acc[t][j] = 0.f;

  float  aReg[16];
  __bf16 bReg[16];

  auto loadTile = [&](int k0) {
    if (aRowOk && (k0 + 32 <= K)) {
      const float4* s = (const float4*)(A + (size_t)(tileM + arow) * K + (k0 + acol));
      float4 f0 = s[0], f1 = s[1], f2 = s[2], f3 = s[3];
      aReg[0] = f0.x; aReg[1] = f0.y; aReg[2]  = f0.z; aReg[3]  = f0.w;
      aReg[4] = f1.x; aReg[5] = f1.y; aReg[6]  = f1.z; aReg[7]  = f1.w;
      aReg[8] = f2.x; aReg[9] = f2.y; aReg[10] = f2.z; aReg[11] = f2.w;
      aReg[12] = f3.x; aReg[13] = f3.y; aReg[14] = f3.z; aReg[15] = f3.w;
    } else {
#pragma unroll
      for (int i = 0; i < 16; ++i) {
        int gr = tileM + arow, gk = k0 + acol + i;
        aReg[i] = (gr < M && gk < K) ? A[(size_t)gr * K + gk] : 0.f;
      }
    }
    if (bColOk && (k0 + 32 <= K)) {
      const uint4* s = (const uint4*)(B + (size_t)(k0 + brow) * N + (tileN + bcol));
      union { uint4 u[2]; __bf16 h[16]; } bb;
      bb.u[0] = s[0]; bb.u[1] = s[1];
#pragma unroll
      for (int i = 0; i < 16; ++i) bReg[i] = bb.h[i];
    } else {
#pragma unroll
      for (int i = 0; i < 16; ++i) {
        int gk = k0 + brow, gc = tileN + bcol + i;
        bReg[i] = (gk < K && gc < N) ? B[(size_t)gk * N + gc] : (__bf16)0.f;
      }
    }
  };

  auto storeTile = [&](int buf) {
    union { uint4 u[2]; __bf16 h[16]; } ap;
#pragma unroll
    for (int i = 0; i < 16; ++i) ap.h[i] = (__bf16)aReg[i];
    *(uint4*)&As[buf][arow][acol]     = ap.u[0];
    *(uint4*)&As[buf][arow][acol + 8] = ap.u[1];
#pragma unroll
    for (int i = 0; i < 16; ++i) BsT[buf][bcol + i][brow] = bReg[i];
  };

  auto compute = [&](int buf) {
    v16bf a;
    const int ar = (wave << 4) + ln;
#pragma unroll
    for (int i = 0; i < 8; ++i) {
      a[i]     = As[buf][ar][(half << 3) + i];
      a[i + 8] = As[buf][ar][16 + (half << 3) + i];
    }
#pragma unroll
    for (int t = 0; t < 4; ++t) {
      v16bf bf;
      const int bc = (t << 4) + ln;
#pragma unroll
      for (int i = 0; i < 8; ++i) {
        bf[i]     = BsT[buf][bc][(half << 3) + i];
        bf[i + 8] = BsT[buf][bc][16 + (half << 3) + i];
      }
      acc[t] = __builtin_amdgcn_wmma_f32_16x16x32_bf16(
          false, a, false, bf, (short)0, acc[t], false, false);
    }
  };

  const int kIter = (K + 31) >> 5;
  loadTile(0);
  storeTile(0);
  __syncthreads();
  for (int ki = 0; ki < kIter; ++ki) {
    const int cur = ki & 1;
    if (ki + 1 < kIter) loadTile((ki + 1) << 5);  // overlap global latency
    compute(cur);
    if (ki + 1 < kIter) storeTile(1 - cur);
    __syncthreads();
  }

  // Epilogue. C/D layout: VGPR j, lane -> row = tileM + 16*wave + 8*half + j,
  // col = tileN + 16*t + ln.
  const int rowB = tileM + (wave << 4) + (half << 3);
  if (mode == 0) {
#pragma unroll
    for (int t = 0; t < 4; ++t) {
      int col = tileN + (t << 4) + ln;
#pragma unroll
      for (int j = 0; j < 8; ++j) {
        int row = rowB + j;
        if (row < M && col < N) {
          float v = acc[t][j];
          if (bias) v += bias[row];
          D[(size_t)row * N + col] = v;
        }
      }
    }
  } else if (mode == 1) {
#pragma unroll
    for (int j = 0; j < 8; ++j) {
      int row = rowB + j;
#pragma unroll
      for (int t = 0; t < 4; ++t) {
        float v = acc[t][j];
        float s = v, q = v * v;
        s += __shfl_xor(s, 1, 32); q += __shfl_xor(q, 1, 32);
        s += __shfl_xor(s, 2, 32); q += __shfl_xor(q, 2, 32);
        s += __shfl_xor(s, 4, 32); q += __shfl_xor(q, 4, 32);
        s += __shfl_xor(s, 8, 32); q += __shfl_xor(q, 8, 32);
        if (ln == 0 && row < M) {
          atomicAdd(&statSum[row], s);
          atomicAdd(&statSsq[row], q);
        }
      }
    }
  } else if (mode == 2) {
#pragma unroll
    for (int j = 0; j < 8; ++j) {
      int row = rowB + j;
      float sc = (row < M) ? scaleV[row] : 0.f;
      float sh = (row < M) ? shiftV[row] : 0.f;
#pragma unroll
      for (int t = 0; t < 4; ++t) {
        int n = (tileN + (t << 4)) >> 4;  // 16-col group == one point's 16 neighbors
        float v = sc * acc[t][j] + sh;
        v = (v >= 0.f) ? v : 0.2f * v;
        v = fmaxf(v, __shfl_xor(v, 1, 32));
        v = fmaxf(v, __shfl_xor(v, 2, 32));
        v = fmaxf(v, __shfl_xor(v, 4, 32));
        v = fmaxf(v, __shfl_xor(v, 8, 32));
        if (ln == 0 && row < M)
          gout[((size_t)batch * M + row) * gldN + n] = v;
      }
    }
  } else {  // mode 3: bn5 + leaky + fused max/mean pooling over n
#pragma unroll
    for (int j = 0; j < 8; ++j) {
      int row = rowB + j;
      float sc = (row < M) ? scaleV[row] : 0.f;
      float sh = (row < M) ? shiftV[row] : 0.f;
#pragma unroll
      for (int t = 0; t < 4; ++t) {
        float v = sc * acc[t][j] + sh;
        v = (v >= 0.f) ? v : 0.2f * v;
        float mx = v, sm = v;
        mx = fmaxf(mx, __shfl_xor(mx, 1, 32)); sm += __shfl_xor(sm, 1, 32);
        mx = fmaxf(mx, __shfl_xor(mx, 2, 32)); sm += __shfl_xor(sm, 2, 32);
        mx = fmaxf(mx, __shfl_xor(mx, 4, 32)); sm += __shfl_xor(sm, 4, 32);
        mx = fmaxf(mx, __shfl_xor(mx, 8, 32)); sm += __shfl_xor(sm, 8, 32);
        if (ln == 0 && row < M) {
          atomicMaxFloat(&pool[(size_t)batch * 2048 + row], mx);
          atomicAdd(&pool[(size_t)batch * 2048 + 1024 + row], sm * (1.f / 2048.f));
        }
      }
    }
  }
}

// ---------------------------------------------------------------------------
// Small helper kernels
// ---------------------------------------------------------------------------
__global__ __launch_bounds__(256)
void fill_kernel(float* __restrict__ p, float v, int n) {
  int t = blockIdx.x * blockDim.x + threadIdx.x;
  if (t < n) p[t] = v;
}

__global__ __launch_bounds__(256)
void finalize_bn_kernel(const float* __restrict__ sum, const float* __restrict__ ssq,
                        const float* __restrict__ g, const float* __restrict__ bb,
                        float* __restrict__ scale, float* __restrict__ shift,
                        int M, float invCnt) {
  int ch = blockIdx.x * blockDim.x + threadIdx.x;
  if (ch >= M) return;
  float mean = sum[ch] * invCnt;
  float var  = ssq[ch] * invCnt - mean * mean;
  float s = g[ch] / sqrtf(var + 1e-5f);
  scale[ch] = s;
  shift[ch] = bb[ch] - mean * s;
}

__global__ __launch_bounds__(256)
void init_pool_kernel(float* __restrict__ pool) {
  int t = blockIdx.x * blockDim.x + threadIdx.x;
  if (t < 8 * 2048) pool[t] = ((t & 2047) < 1024) ? -3.0e38f : 0.f;
}

__global__ __launch_bounds__(256)
void build_poolB_kernel(const float* __restrict__ pool, __bf16* __restrict__ outB) {
  int t = blockIdx.x * blockDim.x + threadIdx.x;   // 2048*16
  if (t >= 2048 * 16) return;
  int k = t >> 4, j = t & 15;
  float v = (j < 8) ? pool[(size_t)j * 2048 + k] : 0.f;
  outB[t] = (__bf16)v;
}

__global__ __launch_bounds__(256)
void build_h_kernel(const float* __restrict__ x, const float* __restrict__ flT,
                    __bf16* __restrict__ hB) {
  int t = blockIdx.x * blockDim.x + threadIdx.x;   // 57600*16
  if (t >= 57600 * 16) return;
  int k = t >> 4, j = t & 15;
  float v = 0.f;
  if (j < 8)
    v = (k < 28800) ? x[(size_t)j * 28800 + k] : flT[(size_t)(k - 28800) * 16 + j];
  hB[t] = (__bf16)v;
}

__global__ __launch_bounds__(256)
void cvt_f32_bf16_kernel(const float* __restrict__ src, __bf16* __restrict__ dst, int n) {
  int t = blockIdx.x * blockDim.x + threadIdx.x;
  if (t < n) dst[t] = (__bf16)src[t];
}

__global__ __launch_bounds__(128)
void copy_emb_kernel(const float* __restrict__ embT, float* __restrict__ eOut) {
  int t = threadIdx.x;
  if (t < 80) {
    int b = t / 10, e = t % 10;
    eOut[t] = embT[e * 16 + b];
  }
}

__global__ __launch_bounds__(256)
void cluster_d_kernel(const float* __restrict__ embT, const float* __restrict__ clu,
                      float* __restrict__ dbuf, float* __restrict__ dcopy) {
  int c = blockIdx.x * blockDim.x + threadIdx.x;
  int b = blockIdx.y;
  if (c >= 800) return;
  float s = 0.f;
#pragma unroll
  for (int e = 0; e < 10; ++e) {
    float df = embT[e * 16 + b] - clu[c * 10 + e];
    s += df * df;
  }
  dbuf[b * 800 + c] = s;
  if (dcopy) dcopy[b * 800 + c] = s;
}

__global__ __launch_bounds__(256)
void q_kernel(const float* __restrict__ d, float* __restrict__ q) {
  __shared__ float red[256];
  int b = blockIdx.x;
  float part = 0.f;
  for (int c = threadIdx.x; c < 800; c += 256)
    part += 1.f / (1.f + d[b * 800 + c]);
  red[threadIdx.x] = part;
  __syncthreads();
  for (int s = 128; s > 0; s >>= 1) {
    if (threadIdx.x < s) red[threadIdx.x] += red[threadIdx.x + s];
    __syncthreads();
  }
  float tot = red[0];
  for (int c = threadIdx.x; c < 800; c += 256)
    q[b * 800 + c] = (1.f / (1.f + d[b * 800 + c])) / tot;
}

__global__ __launch_bounds__(32)
void sim_kernel(const float* __restrict__ e1, const float* __restrict__ e2,
                float* __restrict__ simOut) {
  int b = threadIdx.x;
  if (b < 8) {
    float s = 0.f;
    for (int e = 0; e < 10; ++e) {
      float df = e1[b * 10 + e] - e2[b * 10 + e] + 1e-6f;
      s += df * df;
    }
    simOut[b] = sqrtf(s);
  }
}

// ---------------------------------------------------------------------------
// Host orchestration
// ---------------------------------------------------------------------------
extern "C" void kernel_launch(void* const* d_in, const int* in_sizes, int n_in,
                              void* d_out, int out_size, void* d_ws, size_t ws_size,
                              hipStream_t stream) {
  (void)in_sizes; (void)n_in; (void)out_size; (void)ws_size;

  // ---- input pointers (setup_inputs dict order; params in make_params order)
  const float* img1    = (const float*)d_in[0];
  const float* img2    = (const float*)d_in[1];
  const float* f1      = (const float*)d_in[2];
  const float* f2      = (const float*)d_in[3];
  const float* conv1_w = (const float*)d_in[4];
  const float* conv1_b = (const float*)d_in[5];
  const float* conv2_w = (const float*)d_in[6];
  const float* conv2_b = (const float*)d_in[7];
  const float* conv3_w = (const float*)d_in[8];
  const float* conv3_b = (const float*)d_in[9];
  const float* c1g_w   = (const float*)d_in[10];
  const float* c2g_w   = (const float*)d_in[11];
  const float* c3g_w   = (const float*)d_in[12];
  const float* c4g_w   = (const float*)d_in[13];
  const float* c5g_w   = (const float*)d_in[14];
  const float* bn_g[4] = {(const float*)d_in[15], (const float*)d_in[17],
                          (const float*)d_in[19], (const float*)d_in[21]};
  const float* bn_b[4] = {(const float*)d_in[16], (const float*)d_in[18],
                          (const float*)d_in[20], (const float*)d_in[22]};
  const float* bn5_g   = (const float*)d_in[23];
  const float* bn5_b   = (const float*)d_in[24];
  const float* lin_w   = (const float*)d_in[25];
  const float* lin_b   = (const float*)d_in[26];
  const float* lin1_w  = (const float*)d_in[27];
  const float* lin1_b  = (const float*)d_in[28];
  const float* emb_w   = (const float*)d_in[29];
  const float* emb_b   = (const float*)d_in[30];
  const float* clu_w   = (const float*)d_in[31];

  float* out = (float*)d_out;

  // ---- workspace carve-up (256B aligned)
  char* wsb = (char*)d_ws;
  size_t off = 0;
  auto alloc = [&](size_t bytes) -> void* {
    off = (off + 255) & ~(size_t)255;
    void* p = wsb + off;
    off += bytes;
    return p;
  };
  float*  t1      = (float*)alloc((size_t)8 * 32 * 64 * 64 * 4);
  float*  t2      = (float*)alloc((size_t)8 * 64 * 32 * 32 * 4);
  float*  xbuf    = (float*)alloc((size_t)8 * 28800 * 4);
  float*  xxbuf   = (float*)alloc((size_t)8 * 2048 * 4);
  int*    idxbuf  = (int*)  alloc((size_t)8 * 2048 * 16 * 4);
  float*  g1      = (float*)alloc((size_t)8 * 64  * 2048 * 4);
  float*  g2      = (float*)alloc((size_t)8 * 64  * 2048 * 4);
  float*  g3      = (float*)alloc((size_t)8 * 128 * 2048 * 4);
  float*  g4      = (float*)alloc((size_t)8 * 256 * 2048 * 4);
  __bf16* featB   = (__bf16*)alloc((size_t)512 * 32768 * 2);
  float*  flT     = (float*)alloc((size_t)28800 * 16 * 4);
  float*  hT      = (float*)alloc((size_t)512 * 16 * 4);
  float*  embT1   = (float*)alloc((size_t)16 * 16 * 4);
  float*  embT2   = (float*)alloc((size_t)16 * 16 * 4);
  float*  statSum = (float*)alloc(1024 * 4);
  float*  statSsq = (float*)alloc(1024 * 4);
  float*  scaleB  = (float*)alloc(1024 * 4);
  float*  shiftB  = (float*)alloc(1024 * 4);
  float*  pool    = (float*)alloc((size_t)8 * 2048 * 4);
  float*  dbuf    = (float*)alloc((size_t)8 * 800 * 4);

  auto gemm = [&](const float* A, const __bf16* B, int M, int N, int K, int mode,
                  float* D, const float* bias, float* gout, int batch, int gldN,
                  float* poolp) {
    dim3 g(cdiv_h(N, 64), cdiv_h(M, 64));
    gemm_bf16_wmma<<<g, 128, 0, stream>>>(A, B, M, N, K, mode, D, bias,
                                          statSum, statSsq, scaleB, shiftB,
                                          gout, batch, gldN, poolp);
  };

  const float* Wg[4]   = {c1g_w, c2g_w, c3g_w, c4g_w};
  float*       gbuf[4] = {g1, g2, g3, g4};
  const int    Couts[4] = {64, 64, 128, 256};

  auto run_image = [&](const float* img, const float* f, float* q_out,
                       float* d1_out, float* e_out, float* embT) {
    // CNN branch
    conv2d_leaky_kernel<<<cdiv_h(8 * 32 * 64 * 64, 256), 256, 0, stream>>>(
        img, conv1_w, conv1_b, t1, 8, 3, 128, 128, 32, 5, 5, 2, 2, 64, 64, 0.01f);
    conv2d_leaky_kernel<<<cdiv_h(8 * 64 * 32 * 32, 256), 256, 0, stream>>>(
        t1, conv2_w, conv2_b, t2, 8, 32, 64, 64, 64, 5, 5, 2, 2, 32, 32, 0.01f);
    conv2d_leaky_kernel<<<cdiv_h(8 * 128 * 15 * 15, 256), 256, 0, stream>>>(
        t2, conv3_w, conv3_b, xbuf, 8, 64, 32, 32, 128, 3, 3, 2, 0, 15, 15, 0.01f);

    // DGCNN branch
    const float* fin = f;
    int Cin = 3;
    for (int blk = 0; blk < 4; ++blk) {
      int Cout = Couts[blk];
      int K2 = 2 * Cin;
      xx_kernel<<<cdiv_h(8 * 2048, 256), 256, 0, stream>>>(fin, xxbuf, 8, Cin, 2048);
      topk_kernel<<<dim3(2048 / 32, 8), 32, 0, stream>>>(fin, xxbuf, idxbuf, Cin, 2048);
      fill_kernel<<<cdiv_h(1024, 256), 256, 0, stream>>>(statSum, 0.f, 1024);
      fill_kernel<<<cdiv_h(1024, 256), 256, 0, stream>>>(statSsq, 0.f, 1024);
      // pass 1: BN statistics
      for (int b = 0; b < 8; ++b) {
        build_feat_kernel<<<dim3(128, K2), 256, 0, stream>>>(fin, idxbuf, featB, b, Cin, 2048);
        gemm(Wg[blk], featB, Cout, 32768, K2, 1, nullptr, nullptr, nullptr, b, 0, nullptr);
      }
      finalize_bn_kernel<<<cdiv_h(Cout, 256), 256, 0, stream>>>(
          statSum, statSsq, bn_g[blk], bn_b[blk], scaleB, shiftB, Cout,
          1.f / (8.f * 2048.f * 16.f));
      // pass 2: BN apply + leaky + max over k (fused epilogue)
      for (int b = 0; b < 8; ++b) {
        build_feat_kernel<<<dim3(128, K2), 256, 0, stream>>>(fin, idxbuf, featB, b, Cin, 2048);
        gemm(Wg[blk], featB, Cout, 32768, K2, 2, nullptr, nullptr, gbuf[blk], b, 2048, nullptr);
      }
      fin = gbuf[blk];
      Cin = Cout;
    }

    // c5g: y = fc @ c5g_w^T, bn5 over (b,n), then max/mean pool
    fill_kernel<<<cdiv_h(1024, 256), 256, 0, stream>>>(statSum, 0.f, 1024);
    fill_kernel<<<cdiv_h(1024, 256), 256, 0, stream>>>(statSsq, 0.f, 1024);
    for (int b = 0; b < 8; ++b) {
      build_fc_kernel<<<dim3(8, 512), 256, 0, stream>>>(g1, g2, g3, g4, featB, b);
      gemm(c5g_w, featB, 1024, 2048, 512, 1, nullptr, nullptr, nullptr, b, 0, nullptr);
    }
    finalize_bn_kernel<<<cdiv_h(1024, 256), 256, 0, stream>>>(
        statSum, statSsq, bn5_g, bn5_b, scaleB, shiftB, 1024, 1.f / (8.f * 2048.f));
    init_pool_kernel<<<cdiv_h(8 * 2048, 256), 256, 0, stream>>>(pool);
    for (int b = 0; b < 8; ++b) {
      build_fc_kernel<<<dim3(8, 512), 256, 0, stream>>>(g1, g2, g3, g4, featB, b);
      gemm(c5g_w, featB, 1024, 2048, 512, 3, nullptr, nullptr, nullptr, b, 0, pool);
    }

    // fl = fpool @ lin_w^T + lin_b   (D = flT: 28800 x 16)
    build_poolB_kernel<<<cdiv_h(2048 * 16, 256), 256, 0, stream>>>(pool, featB);
    gemm(lin_w, featB, 28800, 16, 2048, 0, flT, lin_b, nullptr, 0, 0, nullptr);

    // h = concat(x, fl); hT = lin1(h)
    build_h_kernel<<<cdiv_h(57600 * 16, 256), 256, 0, stream>>>(xbuf, flT, featB);
    gemm(lin1_w, featB, 512, 16, 57600, 0, hT, lin1_b, nullptr, 0, 0, nullptr);

    // emb = h @ emb_w^T + emb_b
    cvt_f32_bf16_kernel<<<cdiv_h(512 * 16, 256), 256, 0, stream>>>(hT, featB, 512 * 16);
    gemm(emb_w, featB, 10, 16, 512, 0, embT, emb_b, nullptr, 0, 0, nullptr);
    copy_emb_kernel<<<1, 128, 0, stream>>>(embT, e_out);

    // cluster soft assignment
    cluster_d_kernel<<<dim3(4, 8), 256, 0, stream>>>(embT, clu_w, dbuf, d1_out);
    q_kernel<<<8, 256, 0, stream>>>(dbuf, q_out);
  };

  // outputs: sim[8], q1[6400], q2[6400], e1[80], e2[80], d1[6400]
  run_image(img1, f1, out + 8,    out + 12968, out + 12808, embT1);
  run_image(img2, f2, out + 6408, nullptr,     out + 12888, embT2);
  sim_kernel<<<1, 32, 0, stream>>>(out + 12808, out + 12888, out);
}
